// TransformerDecoderLayer_88158498718390
// MI455X (gfx1250) — matrix operand order repaired and checked by reading, not compiled, not run on hardware
//
#include <hip/hip_runtime.h>
#include <math.h>

// ---------------------------------------------------------------------------
// Types for CDNA5 WMMA (wave32): V_WMMA_F32_16X16X32_BF16
// ---------------------------------------------------------------------------
typedef __bf16 bf16;
typedef __attribute__((ext_vector_type(16))) __bf16 v16bf;
typedef __attribute__((ext_vector_type(8)))  float  v8f;
typedef int v4i __attribute__((vector_size(16)));

__device__ __forceinline__ v8f wmma_bf16(v16bf a, v16bf b, v8f c) {
  return __builtin_amdgcn_wmma_f32_16x16x32_bf16(false, a, false, b, (short)0, c,
                                                 false, false);
}

// ---------------------------------------------------------------------------
// gfx1250 async global->LDS copy (ASYNCcnt path), guarded so compilation
// cannot break if this toolchain lacks the builtin.
// ---------------------------------------------------------------------------
#if defined(__gfx1250__) && __has_builtin(__builtin_amdgcn_global_load_async_to_lds_b128) && __has_builtin(__builtin_amdgcn_s_wait_asynccnt)
#define ASYNC_LDS 1
#else
#define ASYNC_LDS 0
#endif

__device__ __forceinline__ void cp16(bf16* l, const bf16* g) {
#if ASYNC_LDS
  typedef __attribute__((address_space(1))) v4i* gp_t;  // global v4i*
  typedef __attribute__((address_space(3))) v4i* lp_t;  // LDS v4i*
  __builtin_amdgcn_global_load_async_to_lds_b128((gp_t)(void*)g, (lp_t)(void*)l,
                                                 0, 0);
#else
  *(uint4*)(void*)l = *(const uint4*)(const void*)g;
#endif
}
__device__ __forceinline__ void async_wait0() {
#if ASYNC_LDS
  __builtin_amdgcn_s_wait_asynccnt(0);
#endif
}

// ---------------------------------------------------------------------------
// Problem constants
// ---------------------------------------------------------------------------
#define D_MODEL 1024
#define N_HEAD  16
#define HEAD_DIM 64
#define SEQ     2048
#define BATCH   2
#define NTOK    4096
#define E_EXP   8
#define DFF     4096
#define CAP     2048          // int(2.0 * 2 * 4096 / 8)

// ---------------------------------------------------------------------------
// Utility kernels
// ---------------------------------------------------------------------------
__global__ void zero_kernel(float* __restrict__ p, size_t n) {
  size_t i = (size_t)blockIdx.x * blockDim.x + threadIdx.x;
  size_t stride = (size_t)gridDim.x * blockDim.x;
  for (; i < n; i += stride) p[i] = 0.0f;
}

__global__ void cvt_kernel(const float* __restrict__ s, bf16* __restrict__ d, size_t n) {
  size_t i = (size_t)blockIdx.x * blockDim.x + threadIdx.x;
  size_t stride = (size_t)gridDim.x * blockDim.x;
  for (; i < n; i += stride) d[i] = (bf16)s[i];
}

// ---------------------------------------------------------------------------
// Tiled WMMA GEMM with async-LDS double buffering.
//   C[M,N] = act( A[M,K] * B + bias[N] )
//   BT=true : B stored row-major (N,K)  (y = x W^T)
//   BT=false: B stored row-major (K,N)
// A,B are bf16 in global (16B-aligned rows); tiles stream global->LDS via
// GLOBAL_LOAD_ASYNC_TO_LDS_B128 into the idle buffer while WMMA consumes the
// other. Tile 128x64x32; 256 threads = 8 waves; wave -> 16x64 of C.
// M%128==0, N%64==0, K%32==0 at all call sites.
// ---------------------------------------------------------------------------
template<bool BT, bool GELU_ACT, bool OUT_BF>
__global__ __launch_bounds__(256)
void gemm_wmma(const bf16* __restrict__ A, int lda,
               const bf16* __restrict__ Bm, int ldb,
               const float* __restrict__ bias,
               void* __restrict__ Cout, int ldc,
               int M, int N, int K) {
  __shared__ bf16 As[2][128][32];      // [buf][m][k]
  __shared__ bf16 Bsm[2][64 * 32];     // BT: [n][k] ; !BT: [k][n]

  const int tid  = threadIdx.x;
  const int lane = tid & 31;
  const int wid  = tid >> 5;
  const int hi   = lane >> 4;
  const int l16  = lane & 15;
  const int m0 = blockIdx.y * 128;
  const int n0 = blockIdx.x * 64;

  auto stage = [&](int bufi, int kk) {
    // A tile: 128 rows x 4 segments of 8 bf16 -> 512 b128 copies
#pragma unroll
    for (int i = tid; i < 512; i += 256) {
      const int r = i >> 2, s8 = (i & 3) << 3;
      cp16(&As[bufi][r][s8], &A[(size_t)(m0 + r) * lda + kk + s8]);
    }
    if (BT) {                          // (N,K): 64 rows x 4 segs
      const int n = tid >> 2, s8 = (tid & 3) << 3;
      cp16(&Bsm[bufi][n * 32 + s8], &Bm[(size_t)(n0 + n) * ldb + kk + s8]);
    } else {                           // (K,N): 32 rows x 8 segs
      const int kr = tid >> 3, s8 = (tid & 7) << 3;
      cp16(&Bsm[bufi][kr * 64 + s8], &Bm[(size_t)(kk + kr) * ldb + n0 + s8]);
    }
  };

  v8f acc[4];
#pragma unroll
  for (int nb = 0; nb < 4; ++nb)
#pragma unroll
    for (int r = 0; r < 8; ++r) acc[nb][r] = 0.0f;

  stage(0, 0);
  async_wait0();

  const int nk = K >> 5;
  for (int it = 0; it < nk; ++it) {
    __syncthreads();                   // all reads of buf[cur] from it-1 done
    const int cur = it & 1;
    if (it + 1 < nk) stage(cur ^ 1, (it + 1) << 5);  // overlap with WMMA

    v16bf a;
    const int ar = wid * 16 + l16;
    const int kb8 = hi * 16;
#pragma unroll
    for (int j = 0; j < 8; ++j) {
      a[j]     = As[cur][ar][kb8 + j];
      a[j + 8] = As[cur][ar][kb8 + 8 + j];
    }
    const int kb16 = hi * 16;
#pragma unroll
    for (int nb = 0; nb < 4; ++nb) {
      v16bf b;
      const int col = nb * 16 + l16;
#pragma unroll
      for (int j = 0; j < 16; ++j)
        b[j] = BT ? Bsm[cur][col * 32 + kb16 + j]
                  : Bsm[cur][(kb16 + j) * 64 + col];
      acc[nb] = wmma_bf16(a, b, acc[nb]);
    }
    async_wait0();                     // next tile resident before barrier
  }

#pragma unroll
  for (int nb = 0; nb < 4; ++nb) {
    const int cn = n0 + nb * 16 + l16;
    const float bv = bias ? bias[cn] : 0.0f;
#pragma unroll
    for (int r = 0; r < 8; ++r) {
      const int cm = m0 + wid * 16 + r + 8 * hi;
      float v = acc[nb][r] + bv;
      if (GELU_ACT) v = 0.5f * v * (1.0f + erff(v * 0.70710678118654752f));
      if (cm < M && cn < N) {
        if (OUT_BF) ((bf16*)Cout)[(size_t)cm * ldc + cn] = (bf16)v;
        else        ((float*)Cout)[(size_t)cm * ldc + cn] = v;
      }
    }
  }
}

// ---------------------------------------------------------------------------
// Flash attention, bf16 I/O, fp32 online softmax, async double-buffered K/V.
// Grid: (Sq/64, BATCH*N_HEAD). Block: 128 = 4 waves; wave -> 16 q rows.
// ---------------------------------------------------------------------------
__global__ __launch_bounds__(128)
void flash_attn(const bf16* __restrict__ qkv, int ldq,
                int qoff, int koff, int voff,
                bf16* __restrict__ out, int ldo, int Sk) {
  __shared__ bf16 Qs[64][64];
  __shared__ bf16 Ks[2][32][64];
  __shared__ bf16 Vs[2][32][64];
  __shared__ bf16 Ps[4][16][32];

  const int tid  = threadIdx.x;
  const int lane = tid & 31;
  const int wid  = tid >> 5;
  const int hi   = lane >> 4;
  const int l16  = lane & 15;
  const int b = blockIdx.y % BATCH;
  const int h = blockIdx.y / BATCH;
  const int sq0 = blockIdx.x * 64;
  const int hcol = h * HEAD_DIM;
  const float scale = 0.125f;

  auto stageKV = [&](int bufi, int kt) {
#pragma unroll
    for (int i = tid; i < 512; i += 128) {
      const int r = (i >> 3) & 31, s8 = (i & 7) << 3;
      const size_t trow = (size_t)((kt + r) * BATCH + b) * ldq + hcol;
      if (i < 256) cp16(&Ks[bufi][r][s8], &qkv[trow + koff + s8]);
      else         cp16(&Vs[bufi][r][s8], &qkv[trow + voff + s8]);
    }
  };

  // Q block (64x64) + first K/V chunk
#pragma unroll
  for (int i = tid; i < 512; i += 128) {
    const int r = i >> 3, s8 = (i & 7) << 3;
    cp16(&Qs[r][s8],
         &qkv[(size_t)((sq0 + r) * BATCH + b) * ldq + qoff + hcol + s8]);
  }
  stageKV(0, 0);
  async_wait0();
  __syncthreads();

  v16bf a0, a1;
  {
    const int qr = wid * 16 + l16;
    const int kb8 = hi * 16;
#pragma unroll
    for (int j = 0; j < 8; ++j) {
      a0[j]     = Qs[qr][kb8 + j];
      a0[j + 8] = Qs[qr][kb8 + 8 + j];
      a1[j]     = Qs[qr][32 + kb8 + j];
      a1[j + 8] = Qs[qr][32 + kb8 + 8 + j];
    }
  }

  v8f acc[4];
#pragma unroll
  for (int nb = 0; nb < 4; ++nb)
#pragma unroll
    for (int r = 0; r < 8; ++r) acc[nb][r] = 0.0f;
  float mi[8], li[8];
#pragma unroll
  for (int r = 0; r < 8; ++r) { mi[r] = -3.0e38f; li[r] = 0.0f; }

  const int nk = Sk >> 5;
  for (int it = 0; it < nk; ++it) {
    __syncthreads();
    const int cur = it & 1;
    if (it + 1 < nk) stageKV(cur ^ 1, (it + 1) << 5);

    // ---- S = Q K^T over this 32-key chunk (two 16-key tiles) ----
    v8f s0, s1;
#pragma unroll
    for (int r = 0; r < 8; ++r) { s0[r] = 0.0f; s1[r] = 0.0f; }
    {
      v16bf bk;
      const int kb16 = hi * 16;
#pragma unroll
      for (int j = 0; j < 16; ++j) bk[j] = Ks[cur][l16][kb16 + j];
      s0 = wmma_bf16(a0, bk, s0);
#pragma unroll
      for (int j = 0; j < 16; ++j) bk[j] = Ks[cur][l16][32 + kb16 + j];
      s0 = wmma_bf16(a1, bk, s0);
#pragma unroll
      for (int j = 0; j < 16; ++j) bk[j] = Ks[cur][16 + l16][kb16 + j];
      s1 = wmma_bf16(a0, bk, s1);
#pragma unroll
      for (int j = 0; j < 16; ++j) bk[j] = Ks[cur][16 + l16][32 + kb16 + j];
      s1 = wmma_bf16(a1, bk, s1);
    }

    // ---- online softmax (row reductions within 16-lane groups) ----
    float p0[8], p1[8], al[8];
#pragma unroll
    for (int r = 0; r < 8; ++r) {
      float v0 = s0[r] * scale, v1 = s1[r] * scale;
      float mx = fmaxf(v0, v1);
#pragma unroll
      for (int msk = 8; msk >= 1; msk >>= 1)
        mx = fmaxf(mx, __shfl_xor(mx, msk, 32));
      float mn = fmaxf(mi[r], mx);
      al[r] = __expf(mi[r] - mn);
      p0[r] = __expf(v0 - mn);
      p1[r] = __expf(v1 - mn);
      float rs = p0[r] + p1[r];
#pragma unroll
      for (int msk = 8; msk >= 1; msk >>= 1)
        rs += __shfl_xor(rs, msk, 32);
      li[r] = li[r] * al[r] + rs;
      mi[r] = mn;
    }
#pragma unroll
    for (int nb = 0; nb < 4; ++nb)
#pragma unroll
      for (int r = 0; r < 8; ++r) acc[nb][r] *= al[r];

    // ---- stage P: C layout -> LDS -> reload in A layout ----
#pragma unroll
    for (int r = 0; r < 8; ++r) {
      const int prow = r + 8 * hi;
      Ps[wid][prow][l16]      = (bf16)p0[r];
      Ps[wid][prow][16 + l16] = (bf16)p1[r];
    }
    __syncthreads();

    v16bf pa;
    {
      const int kb8 = hi * 16;
#pragma unroll
      for (int j = 0; j < 8; ++j) {
        pa[j]     = Ps[wid][l16][kb8 + j];
        pa[j + 8] = Ps[wid][l16][kb8 + 8 + j];
      }
    }
    const int kb16 = hi * 16;
#pragma unroll
    for (int nb = 0; nb < 4; ++nb) {
      v16bf vb;
      const int d = nb * 16 + l16;
#pragma unroll
      for (int j = 0; j < 16; ++j) vb[j] = Vs[cur][kb16 + j][d];
      acc[nb] = wmma_bf16(pa, vb, acc[nb]);
    }
    async_wait0();
  }

#pragma unroll
  for (int r = 0; r < 8; ++r) {
    const float inv = 1.0f / li[r];
    const int s = sq0 + wid * 16 + r + 8 * hi;
    const size_t orow = (size_t)(s * BATCH + b) * ldo + hcol;
#pragma unroll
    for (int nb = 0; nb < 4; ++nb)
      out[orow + nb * 16 + l16] = (bf16)(acc[nb][r] * inv);
  }
}

// ---------------------------------------------------------------------------
// MoE gating / routing / scatter / combine / aux
// ---------------------------------------------------------------------------
__global__ __launch_bounds__(32)
void gate_kernel(const bf16* __restrict__ x, const float* __restrict__ wg,
                 int* __restrict__ idx1, int* __restrict__ idx2,
                 float* __restrict__ g1, float* __restrict__ g2,
                 float* __restrict__ probsum) {
  const int t = blockIdx.x;
  const int lane = threadIdx.x;
  float part[E_EXP];
#pragma unroll
  for (int e = 0; e < E_EXP; ++e) part[e] = 0.0f;
  for (int d = lane; d < D_MODEL; d += 32) {
    const float xv = (float)x[(size_t)t * D_MODEL + d];
#pragma unroll
    for (int e = 0; e < E_EXP; ++e) part[e] += xv * wg[d * E_EXP + e];
  }
#pragma unroll
  for (int e = 0; e < E_EXP; ++e)
#pragma unroll
    for (int msk = 16; msk >= 1; msk >>= 1)
      part[e] += __shfl_xor(part[e], msk, 32);

  float mx = part[0];
#pragma unroll
  for (int e = 1; e < E_EXP; ++e) mx = fmaxf(mx, part[e]);
  float se = 0.0f, p[E_EXP];
#pragma unroll
  for (int e = 0; e < E_EXP; ++e) { p[e] = __expf(part[e] - mx); se += p[e]; }
  const float inv = 1.0f / se;
#pragma unroll
  for (int e = 0; e < E_EXP; ++e) p[e] *= inv;

  int i1 = 0; float b1 = p[0];
#pragma unroll
  for (int e = 1; e < E_EXP; ++e) if (p[e] > b1) { b1 = p[e]; i1 = e; }
  int i2 = (i1 == 0) ? 1 : 0; float b2 = p[i2];
#pragma unroll
  for (int e = 0; e < E_EXP; ++e)
    if (e != i1 && p[e] > b2) { b2 = p[e]; i2 = e; }

  if (lane == 0) {
    idx1[t] = i1; idx2[t] = i2; g1[t] = b1; g2[t] = b2;
#pragma unroll
    for (int e = 0; e < E_EXP; ++e) atomicAdd(&probsum[e], p[e]);
  }
}

__global__ void scan_kernel(const int* __restrict__ idx1, const int* __restrict__ idx2,
                            int* __restrict__ loc1, int* __restrict__ loc2,
                            int* __restrict__ count1, int n) {
  const int e = threadIdx.x;
  if (e >= E_EXP) return;
  int c1 = 0;
  for (int t = 0; t < n; ++t)
    if (idx1[t] == e) loc1[t] = c1++;
  count1[e] = c1;
  int c2 = 0;
  for (int t = 0; t < n; ++t)
    if (idx2[t] == e) loc2[t] = c1 + c2++;
}

__global__ __launch_bounds__(256)
void scatter_kernel(const bf16* __restrict__ x,
                    const int* __restrict__ idx1, const int* __restrict__ idx2,
                    const int* __restrict__ loc1, const int* __restrict__ loc2,
                    bf16* __restrict__ buf) {
  const int t = blockIdx.x;
  const int e1 = idx1[t], e2 = idx2[t];
  const int l1 = loc1[t], l2 = loc2[t];
#pragma unroll
  for (int k = 0; k < 4; ++k) {
    const int d = threadIdx.x + k * 256;
    const bf16 v = x[(size_t)t * D_MODEL + d];
    if (l1 < CAP) buf[((size_t)e1 * CAP + l1) * D_MODEL + d] = v;
    if (l2 < CAP) buf[((size_t)e2 * CAP + l2) * D_MODEL + d] = v;
  }
}

__global__ __launch_bounds__(256)
void combine_kernel(const float* __restrict__ y,
                    const int* __restrict__ idx1, const int* __restrict__ idx2,
                    const int* __restrict__ loc1, const int* __restrict__ loc2,
                    const float* __restrict__ g1, const float* __restrict__ g2,
                    float* __restrict__ out) {
  const int t = blockIdx.x;
  const int e1 = idx1[t], e2 = idx2[t];
  const int l1 = loc1[t], l2 = loc2[t];
  const float w1 = g1[t], w2 = g2[t];
#pragma unroll
  for (int k = 0; k < 4; ++k) {
    const int d = threadIdx.x + k * 256;
    float v = 0.0f;
    if (l1 < CAP) v += w1 * y[((size_t)e1 * CAP + l1) * D_MODEL + d];
    if (l2 < CAP) v += w2 * y[((size_t)e2 * CAP + l2) * D_MODEL + d];
    out[(size_t)t * D_MODEL + d] = v;
  }
}

__global__ void aux_kernel(const int* __restrict__ count1,
                           const float* __restrict__ probsum,
                           float* __restrict__ aux_out) {
  if (threadIdx.x == 0 && blockIdx.x == 0) {
    const float invn = 1.0f / (float)NTOK;
    float s = 0.0f;
    for (int e = 0; e < E_EXP; ++e)
      s += ((float)count1[e] * invn) * (probsum[e] * invn);
    aux_out[0] = 0.01f * (float)E_EXP * s;
  }
}

// ---------------------------------------------------------------------------
// Triple layernorm, one row per workgroup.
// ---------------------------------------------------------------------------
__device__ __forceinline__ float block_sum(float v, float* red) {
  const int tid = threadIdx.x;
  __syncthreads();
  red[tid] = v;
  __syncthreads();
#pragma unroll
  for (int s = 128; s >= 1; s >>= 1) {
    if (tid < s) red[tid] += red[tid + s];
    __syncthreads();
  }
  return red[0];
}

__global__ __launch_bounds__(256)
void ln3_kernel(const float* __restrict__ x,
                const float* __restrict__ g1, const float* __restrict__ b1,
                const float* __restrict__ g2, const float* __restrict__ b2,
                const float* __restrict__ g3, const float* __restrict__ b3,
                float* __restrict__ out) {
  __shared__ float red[256];
  const size_t row = (size_t)blockIdx.x * D_MODEL;
  const int tid = threadIdx.x;
  float v[4];
#pragma unroll
  for (int k = 0; k < 4; ++k) v[k] = x[row + tid + k * 256];

  const float* gs[3] = {g1, g2, g3};
  const float* bs[3] = {b1, b2, b3};
#pragma unroll
  for (int pass = 0; pass < 3; ++pass) {
    float s = 0.0f;
#pragma unroll
    for (int k = 0; k < 4; ++k) s += v[k];
    const float mean = block_sum(s, red) * (1.0f / D_MODEL);
    float sq = 0.0f;
#pragma unroll
    for (int k = 0; k < 4; ++k) { const float d = v[k] - mean; sq += d * d; }
    const float var = block_sum(sq, red) * (1.0f / D_MODEL);
    const float rstd = rsqrtf(var + 1e-5f);
    const float* g = gs[pass]; const float* b = bs[pass];
#pragma unroll
    for (int k = 0; k < 4; ++k) {
      const int d = tid + k * 256;
      v[k] = (v[k] - mean) * rstd * g[d] + b[d];
    }
  }
#pragma unroll
  for (int k = 0; k < 4; ++k) out[row + tid + k * 256] = v[k];
}

// ---------------------------------------------------------------------------
// Host orchestration
// ---------------------------------------------------------------------------
extern "C" void kernel_launch(void* const* d_in, const int* in_sizes, int n_in,
                              void* d_out, int out_size, void* d_ws, size_t ws_size,
                              hipStream_t stream) {
  const float* tgt      = (const float*)d_in[0];
  const float* memory   = (const float*)d_in[1];
  const float* w_qkv_sa = (const float*)d_in[2];
  const float* b_qkv_sa = (const float*)d_in[3];
  const float* w_o_sa   = (const float*)d_in[4];
  const float* b_o_sa   = (const float*)d_in[5];
  const float* w_qkv_ca = (const float*)d_in[6];
  const float* b_qkv_ca = (const float*)d_in[7];
  const float* w_o_ca   = (const float*)d_in[8];
  const float* b_o_ca   = (const float*)d_in[9];
  const float* ln1_g = (const float*)d_in[10];
  const float* ln1_b = (const float*)d_in[11];
  const float* ln2_g = (const float*)d_in[12];
  const float* ln2_b = (const float*)d_in[13];
  const float* ln3_g = (const float*)d_in[14];
  const float* ln3_b = (const float*)d_in[15];
  const float* w_gate = (const float*)d_in[16];
  const float* w1  = (const float*)d_in[17];
  const float* b1e = (const float*)d_in[18];
  const float* w2  = (const float*)d_in[19];
  const float* b2e = (const float*)d_in[20];
  float* out = (float*)d_out;

  // ---- workspace (byte allocator, 256B aligned chunks) ----
  char* wsp = (char*)d_ws;
  auto alloc = [&](size_t bytes) {
    char* p = wsp;
    wsp += (bytes + 255) & ~(size_t)255;
    return p;
  };
  bf16* tgt_bf  = (bf16*)alloc((size_t)NTOK * D_MODEL * 2);
  bf16* mem_bf  = (bf16*)alloc((size_t)NTOK * D_MODEL * 2);
  bf16* wqsa_bf = (bf16*)alloc((size_t)3 * D_MODEL * D_MODEL * 2);
  bf16* wosa_bf = (bf16*)alloc((size_t)D_MODEL * D_MODEL * 2);
  bf16* wqca_bf = (bf16*)alloc((size_t)3 * D_MODEL * D_MODEL * 2);
  bf16* woca_bf = (bf16*)alloc((size_t)D_MODEL * D_MODEL * 2);
  bf16* w1_bf   = (bf16*)alloc((size_t)E_EXP * D_MODEL * DFF * 2);
  bf16* w2_bf   = (bf16*)alloc((size_t)E_EXP * DFF * D_MODEL * 2);
  bf16* qkv_bf  = (bf16*)alloc((size_t)NTOK * 3 * D_MODEL * 2);
  bf16* attn_bf = (bf16*)alloc((size_t)NTOK * D_MODEL * 2);
  bf16* x1_bf   = (bf16*)alloc((size_t)NTOK * D_MODEL * 2);
  bf16* x2_bf   = (bf16*)alloc((size_t)NTOK * D_MODEL * 2);
  bf16* buf_bf  = (bf16*)alloc((size_t)E_EXP * CAP * D_MODEL * 2);
  bf16* h_bf    = (bf16*)alloc((size_t)CAP * DFF * 2);
  float* yall    = (float*)alloc((size_t)E_EXP * CAP * D_MODEL * 4);
  float* moe_out = (float*)alloc((size_t)NTOK * D_MODEL * 4);
  float* probsum = (float*)alloc(8 * 4);
  float* g1buf   = (float*)alloc(NTOK * 4);
  float* g2buf   = (float*)alloc(NTOK * 4);
  int* idx1   = (int*)alloc(NTOK * 4);
  int* idx2   = (int*)alloc(NTOK * 4);
  int* loc1   = (int*)alloc(NTOK * 4);
  int* loc2   = (int*)alloc(NTOK * 4);
  int* count1 = (int*)alloc(E_EXP * 4);

  auto cvt = [&](const float* s, bf16* d, size_t n) {
    cvt_kernel<<<4096, 256, 0, stream>>>(s, d, n);
  };
  auto gemm = [&](bool bt, bool gelu, bool obf,
                  const bf16* A, int lda, const bf16* Bm, int ldb,
                  const float* bias, void* C, int ldc, int M, int N, int K) {
    dim3 grid(N / 64, M / 128), blk(256);
    if (bt) {
      if (gelu) {
        if (obf) gemm_wmma<true, true, true><<<grid, blk, 0, stream>>>(A, lda, Bm, ldb, bias, C, ldc, M, N, K);
        else     gemm_wmma<true, true, false><<<grid, blk, 0, stream>>>(A, lda, Bm, ldb, bias, C, ldc, M, N, K);
      } else {
        if (obf) gemm_wmma<true, false, true><<<grid, blk, 0, stream>>>(A, lda, Bm, ldb, bias, C, ldc, M, N, K);
        else     gemm_wmma<true, false, false><<<grid, blk, 0, stream>>>(A, lda, Bm, ldb, bias, C, ldc, M, N, K);
      }
    } else {
      if (gelu) {
        if (obf) gemm_wmma<false, true, true><<<grid, blk, 0, stream>>>(A, lda, Bm, ldb, bias, C, ldc, M, N, K);
        else     gemm_wmma<false, true, false><<<grid, blk, 0, stream>>>(A, lda, Bm, ldb, bias, C, ldc, M, N, K);
      } else {
        if (obf) gemm_wmma<false, false, true><<<grid, blk, 0, stream>>>(A, lda, Bm, ldb, bias, C, ldc, M, N, K);
        else     gemm_wmma<false, false, false><<<grid, blk, 0, stream>>>(A, lda, Bm, ldb, bias, C, ldc, M, N, K);
      }
    }
  };

  // ---- one-time fp32 -> bf16 conversions (inputs + weights) ----
  cvt(tgt, tgt_bf, (size_t)NTOK * D_MODEL);
  cvt(memory, mem_bf, (size_t)NTOK * D_MODEL);
  cvt(w_qkv_sa, wqsa_bf, (size_t)3 * D_MODEL * D_MODEL);
  cvt(w_o_sa, wosa_bf, (size_t)D_MODEL * D_MODEL);
  cvt(w_qkv_ca, wqca_bf, (size_t)3 * D_MODEL * D_MODEL);
  cvt(w_o_ca, woca_bf, (size_t)D_MODEL * D_MODEL);
  cvt(w1, w1_bf, (size_t)E_EXP * D_MODEL * DFF);
  cvt(w2, w2_bf, (size_t)E_EXP * DFF * D_MODEL);

  // ===== Self-attention =====
  gemm(true, false, true, tgt_bf, D_MODEL, wqsa_bf, D_MODEL, b_qkv_sa,
       qkv_bf, 3 * D_MODEL, NTOK, 3 * D_MODEL, D_MODEL);
  flash_attn<<<dim3(SEQ / 64, BATCH * N_HEAD), 128, 0, stream>>>(
      qkv_bf, 3 * D_MODEL, 0, D_MODEL, 2 * D_MODEL, attn_bf, D_MODEL, SEQ);
  gemm(true, false, true, attn_bf, D_MODEL, wosa_bf, D_MODEL, b_o_sa,
       x1_bf, D_MODEL, NTOK, D_MODEL, D_MODEL);

  // ===== Cross-attention (Q from x1, K/V from memory) =====
  gemm(true, false, true, x1_bf, D_MODEL, wqca_bf, D_MODEL, b_qkv_ca,
       qkv_bf, 3 * D_MODEL, NTOK, D_MODEL, D_MODEL);
  gemm(true, false, true, mem_bf, D_MODEL, wqca_bf + (size_t)D_MODEL * D_MODEL,
       D_MODEL, b_qkv_ca + D_MODEL, qkv_bf + D_MODEL, 3 * D_MODEL,
       NTOK, 2 * D_MODEL, D_MODEL);
  flash_attn<<<dim3(SEQ / 64, BATCH * N_HEAD), 128, 0, stream>>>(
      qkv_bf, 3 * D_MODEL, 0, D_MODEL, 2 * D_MODEL, attn_bf, D_MODEL, SEQ);
  gemm(true, false, true, attn_bf, D_MODEL, woca_bf, D_MODEL, b_o_ca,
       x2_bf, D_MODEL, NTOK, D_MODEL, D_MODEL);

  // ===== MoE =====
  zero_kernel<<<2048, 256, 0, stream>>>((float*)buf_bf,
                                        (size_t)E_EXP * CAP * D_MODEL / 2);
  zero_kernel<<<1, 256, 0, stream>>>(probsum, 8);
  gate_kernel<<<NTOK, 32, 0, stream>>>(x2_bf, w_gate, idx1, idx2, g1buf, g2buf, probsum);
  scan_kernel<<<1, 32, 0, stream>>>(idx1, idx2, loc1, loc2, count1, NTOK);
  scatter_kernel<<<NTOK, 256, 0, stream>>>(x2_bf, idx1, idx2, loc1, loc2, buf_bf);
  for (int e = 0; e < E_EXP; ++e) {
    gemm(false, true, true, buf_bf + (size_t)e * CAP * D_MODEL, D_MODEL,
         w1_bf + (size_t)e * D_MODEL * DFF, DFF, b1e + (size_t)e * DFF,
         h_bf, DFF, CAP, DFF, D_MODEL);
    gemm(false, false, false, h_bf, DFF,
         w2_bf + (size_t)e * DFF * D_MODEL, D_MODEL, b2e + (size_t)e * D_MODEL,
         yall + (size_t)e * CAP * D_MODEL, D_MODEL, CAP, D_MODEL, DFF);
  }
  combine_kernel<<<NTOK, 256, 0, stream>>>(yall, idx1, idx2, loc1, loc2,
                                           g1buf, g2buf, moe_out);
  aux_kernel<<<1, 32, 0, stream>>>(count1, probsum, out + (size_t)NTOK * D_MODEL);

  // ===== 3x LayerNorm =====
  ln3_kernel<<<NTOK, 256, 0, stream>>>(moe_out, ln1_g, ln1_b, ln2_g, ln2_b,
                                       ln3_g, ln3_b, out);
}
